// ProtoLayer_75771813036034
// MI455X (gfx1250) — matrix operand order repaired and checked by reading, not compiled, not run on hardware
//
#include <hip/hip_runtime.h>
#include <hip/hip_bf16.h>
#include <math.h>

typedef __attribute__((ext_vector_type(16))) _Float16 v16h;
typedef __attribute__((ext_vector_type(8)))  _Float16 v8h;
typedef __attribute__((ext_vector_type(8)))  float    v8f;

union Frag16 { v16h v; v8h h[2]; };

// ---------------------------------------------------------------------------
// Kernel 1: 2x2 avg pool + double L2 normalize.
// One block per (b,m) row (4096 rows), 256 threads, 8 channels each.
// Writes fN (f32, f = v/max(||v||,1e-12)) and A (f16, f/na, na=||f||+1e-3).
// ---------------------------------------------------------------------------
__global__ __launch_bounds__(256) void pool_norm_kernel(
    const float* __restrict__ x, float* __restrict__ fN, _Float16* __restrict__ Af) {
  const int row = blockIdx.x;          // b*64 + m
  const int b = row >> 6, m = row & 63;
  const int i = m >> 3, j = m & 7;
  const int tid = threadIdx.x;
  const size_t xb = ((size_t)b * 2048) * 256 + (size_t)(2 * i) * 16 + 2 * j;

  float v[8];
  float s = 0.f;
#pragma unroll
  for (int q = 0; q < 8; ++q) {
    const int c = tid + q * 256;
    const float* p = x + xb + (size_t)c * 256;
    v[q] = 0.25f * (p[0] + p[1] + p[16] + p[17]);
    s += v[q] * v[q];
  }
  // wave32 reduction, then cross-wave via LDS
  for (int off = 16; off; off >>= 1) s += __shfl_xor(s, off, 32);
  __shared__ float red[8];
  if ((tid & 31) == 0) red[tid >> 5] = s;
  __syncthreads();
  const float tot = red[0] + red[1] + red[2] + red[3] + red[4] + red[5] + red[6] + red[7];
  const float n1   = sqrtf(tot);
  const float d    = fmaxf(n1, 1e-12f);
  const float invd = 1.0f / d;
  const float na   = sqrtf(tot * invd * invd) + 1e-3f;  // ||f|| + eps_cos
  const float sA   = invd / na;

  const size_t ro = (size_t)row * 2048;
#pragma unroll
  for (int q = 0; q < 8; ++q) {
    const int c = tid + q * 256;
    fN[ro + c] = v[q] * invd;
    Af[ro + c] = (_Float16)(v[q] * sA);
  }
}

// ---------------------------------------------------------------------------
// Kernel 2: prototype column normalize -> Bt (f16), TRANSPOSED [Npad=1024, K=2048]
// so the GEMM's per-lane B fragment loads are contiguous. Cols 1000..1023 = 0.
// ---------------------------------------------------------------------------
__global__ __launch_bounds__(256) void proto_norm_kernel(
    const float* __restrict__ proto, _Float16* __restrict__ Bt) {
  const int p = blockIdx.x;            // 0..1023
  const int tid = threadIdx.x;
  const size_t bo = (size_t)p * 2048;
  if (p >= 1000) {
#pragma unroll
    for (int q = 0; q < 8; ++q) Bt[bo + tid + q * 256] = (_Float16)0.0f;
    return;
  }
  float v[8];
  float s = 0.f;
#pragma unroll
  for (int q = 0; q < 8; ++q) {
    const int c = tid + q * 256;
    v[q] = proto[(size_t)c * 1000 + p];
    s += v[q] * v[q];
  }
  for (int off = 16; off; off >>= 1) s += __shfl_xor(s, off, 32);
  __shared__ float red[8];
  if ((tid & 31) == 0) red[tid >> 5] = s;
  __syncthreads();
  const float tot = red[0] + red[1] + red[2] + red[3] + red[4] + red[5] + red[6] + red[7];
  const float d    = fmaxf(sqrtf(tot), 1e-12f);
  const float invd = 1.0f / d;
  const float nb   = sqrtf(tot * invd * invd) + 1e-3f;
  const float sc   = invd / nb;
#pragma unroll
  for (int q = 0; q < 8; ++q) {
    const int c = tid + q * 256;
    Bt[bo + c] = (_Float16)(v[q] * sc);
  }
}

// ---------------------------------------------------------------------------
// Kernel 3: cosine GEMM via v_wmma_f32_16x16x32_f16. One wave = one 16x16 tile.
// A [4096,2048] f16 row-major; Bt [1024,2048] f16 (N-major). K loop of 64 WMMAs.
// ISA 16-bit A layout: lanes 0-15 = M, lane-half selects K {0..7,16..23} vs
// {8..15,24..31}; this code loads A and B alike: lane-half selects K 0..15 vs
// 16..31, contiguous. Result transposed through LDS so stores are m-contiguous.
// Writes cosvalue [b][p][m] and cosdist = 1 - cosvalue.
// ---------------------------------------------------------------------------
__global__ __launch_bounds__(256) void gemm_cos_kernel(
    const _Float16* __restrict__ A, const _Float16* __restrict__ Bt,
    float* __restrict__ cosv, float* __restrict__ cosd) {
  __shared__ float s[8][16][17];
  const int tid  = threadIdx.x;
  const int wave = tid >> 5;
  const int lane = tid & 31;
  const int hi   = lane >> 4;          // lane-half
  const int l16  = lane & 15;
  const int rowTile = blockIdx.y;      // 0..255  (16 rows of A each)
  const int colTile = blockIdx.x * 8 + wave;  // 0..63; 63 tiles valid
  if (colTile >= 63) return;           // no __syncthreads below -> safe early exit

  const int b     = rowTile >> 2;
  const int mBase = (rowTile & 3) * 16;
  const int pBase = colTile * 16;

  const _Float16* aRow = A  + (size_t)(rowTile * 16 + l16) * 2048;
  const _Float16* bRow = Bt + (size_t)(pBase + l16) * 2048;

  v8f c = {};
  for (int k0 = 0; k0 < 2048; k0 += 32) {
    Frag16 af, bf;
    af.v    = *(const v16h*)(aRow + k0 + hi * 16);      // K = k0 + hi*16 + 0..15
    bf.v    = *(const v16h*)(bRow + k0 + hi * 16);      // K = k0 + hi*16 + 0..15
    c = __builtin_amdgcn_wmma_f32_16x16x32_f16(
        /*neg_a=*/false, af.v, /*neg_b=*/false, bf.v,
        /*c_mod=*/(short)0, c, /*reuse_a=*/false, /*reuse_b=*/false);
  }

  // C layout: lane L, vgpr r -> M = r + (L>=16)*8, N = L%16. Transpose in LDS.
#pragma unroll
  for (int r = 0; r < 8; ++r) s[wave][r + hi * 8][l16] = c[r];
  asm volatile("s_wait_dscnt 0" ::: "memory");   // wave-local LDS RAW ordering

#pragma unroll
  for (int r = 0; r < 8; ++r) {
    const int pl = r + hi * 8;
    const int p  = pBase + pl;
    if (p < 1000) {
      const float val = s[wave][l16][pl];        // lane varies m -> coalesced
      const size_t o  = (size_t)(b * 1000 + p) * 64 + mBase + l16;
      cosv[o] = val;
      cosd[o] = 1.0f - val;
    }
  }
}

// ---------------------------------------------------------------------------
// Kernel 4: per (b,p): max/argmax over m (first-max tie-break like jnp.argmax),
// sim_r, contrib = sim*weight*sigmoid(weight_e), out = sum over 10 protos.
// 320 blocks x 200 threads; each thread = one (b,p).
// ---------------------------------------------------------------------------
__global__ __launch_bounds__(200) void reduce_kernel(
    const float* __restrict__ cosv, const float* __restrict__ w,
    const float* __restrict__ we, float* __restrict__ outO,
    float* __restrict__ contribO, float* __restrict__ simO,
    float* __restrict__ argF, int* __restrict__ argI) {
  const int blk = blockIdx.x;                  // 0..319
  const int b   = blk / 5;
  const int p   = (blk % 5) * 200 + threadIdx.x;
  const float* cv = cosv + (size_t)(b * 1000 + p) * 64;

  float best = cv[0];
  int   bi   = 0;
  for (int m2 = 1; m2 < 64; ++m2) {
    const float t = cv[m2];
    if (t > best) { best = t; bi = m2; }       // strict > keeps first max
  }
  const int idx = b * 1000 + p;
  simO[idx] = best;
  argF[idx] = (float)bi;
  argI[idx] = bi;
  const float sig = 1.0f / (1.0f + expf(-we[p]));
  const float ctr = best * w[p] * sig;
  contribO[idx] = ctr;

  __shared__ float sh[200];
  sh[threadIdx.x] = ctr;
  __syncthreads();
  if ((threadIdx.x % 10) == 0) {
    float acc = 0.f;
#pragma unroll
    for (int q = 0; q < 10; ++q) acc += sh[threadIdx.x + q];
    outO[b * 100 + p / 10] = acc;
  }
}

// ---------------------------------------------------------------------------
// Kernel 5: maxfs gather: copy fN[b*64+argmax] (2048 f32) to maxfs_u[b][p].
// fN (33 MB) is L2-resident, so this is write-bandwidth bound (~524 MB).
// ---------------------------------------------------------------------------
__global__ __launch_bounds__(256) void gather_kernel(
    const float* __restrict__ fN, const int* __restrict__ argI,
    float* __restrict__ maxfs) {
  const int bp  = blockIdx.x;                  // 0..63999
  const int b   = bp / 1000;
  const int idx = argI[bp];
  const float4* src = (const float4*)(fN + (size_t)(b * 64 + idx) * 2048);
  float4*       dst = (float4*)(maxfs + (size_t)bp * 2048);
  const int tid = threadIdx.x;
  dst[tid]       = src[tid];
  dst[tid + 256] = src[tid + 256];
}

// ---------------------------------------------------------------------------
extern "C" void kernel_launch(void* const* d_in, const int* in_sizes, int n_in,
                              void* d_out, int out_size, void* d_ws, size_t ws_size,
                              hipStream_t stream) {
  const float* x     = (const float*)d_in[0];  // [64,2048,16,16]
  const float* proto = (const float*)d_in[1];  // [1,2048,1000]
  const float* w     = (const float*)d_in[2];  // [100,10]
  const float* we    = (const float*)d_in[3];  // [100,10]
  float* out = (float*)d_out;
  char*  ws  = (char*)d_ws;

  // workspace layout
  float*     fN   = (float*)ws;                         // 33,554,432 B
  _Float16*  Af   = (_Float16*)(ws + 33554432);         // 16,777,216 B
  _Float16*  Bt   = (_Float16*)(ws + 50331648);         //  4,194,304 B
  int*       argI = (int*)(ws + 54525952);              //    256,000 B

  // d_out layout (tuple concatenated, floats)
  float* outO     = out;                  //   6,400
  float* contribO = out + 6400;           //  64,000
  float* simO     = out + 70400;          //  64,000
  float* cosv     = out + 134400;         // 4,096,000
  float* cosd     = out + 4230400;        // 4,096,000
  float* maxfs    = out + 8326400;        // 131,072,000
  float* argF     = out + 139398400;      //  64,000

  pool_norm_kernel <<<4096, 256, 0, stream>>>(x, fN, Af);
  proto_norm_kernel<<<1024, 256, 0, stream>>>(proto, Bt);
  gemm_cos_kernel  <<<dim3(8, 256), 256, 0, stream>>>(Af, Bt, cosv, cosd);
  reduce_kernel    <<<320, 200, 0, stream>>>(cosv, w, we, outO, contribO, simO, argF, argI);
  gather_kernel    <<<64000, 256, 0, stream>>>(fN, argI, maxfs);
}
